// MLPLSTMPolicy_5420248728135
// MI455X (gfx1250) — matrix operand order repaired and checked by reading, compiled, not run on hardware
//
#include <hip/hip_runtime.h>
#include <hip/hip_bf16.h>

// ---------------------------------------------------------------------------
// Fused MLP->LSTM->heads policy forward for MI455X (gfx1250, wave32, WMMA).
// One persistent block per 16 environments; sequential T loop inside block.
// Matmuls: v_wmma_f32_16x16x32_f16 (f32 accumulate). All weight B-fragments
// are explicitly preloaded into SSA registers before the T loop (immune to
// the async-asm memory clobbers), so the steady-state loop is LDS + WMMA only.
// obs[t+1] is double-buffered into LDS via global_load_async_to_lds_b32.
// ---------------------------------------------------------------------------

#define T_   128
#define N_   2048
#define OBS_ 54
#define HID_ 128
#define G_   512   // 4*LSTM_H
#define ACT_ 6

typedef __attribute__((ext_vector_type(16))) _Float16 v16h;
typedef __attribute__((ext_vector_type(8)))  _Float16 v8h;
typedef __attribute__((ext_vector_type(8)))  float    v8f;

#define SXP  136   // padded f16 row stride for 128-wide activations (bank spread)
#define SOP  72    // padded f16 row stride for obs (64 logical cols)
#define SFP  56    // padded f32 row stride for raw obs staging (54 logical)
#define GSTR 516   // padded f32 row stride for 512-wide gates

// ---- A-matrix fragment (16x32 f16, MxK) from an LDS tile -------------------
// ISA layout: lanes 0-15 hold rows M=0..15 with K = {0..7, 16..23};
//             lanes 16-31 same rows with K = {8..15, 24..31}.
__device__ __forceinline__ v16h afrag(const _Float16* X, int stride, int ks, int lane) {
  const _Float16* p = X + (lane & 15) * stride + ks * 32 + ((lane >> 4) * 8);
  v8h lo = *(const v8h*)(p);
  v8h hi = *(const v8h*)(p + 16);
  return __builtin_shufflevector(lo, hi, 0,1,2,3,4,5,6,7,8,9,10,11,12,13,14,15);
}

// ---- B-matrix fragment (32x16 f16, KxN) from row-major f16 weights ---------
// B = W^T: lane l supplies 16 K of column N=l%16 from row (ct*16 + l%16) of W.
// lanes 0-15: K={0..7,16..23}, lanes 16-31: K={8..15,24..31} (matches afrag).
__device__ __forceinline__ v16h bfrag(const _Float16* __restrict__ W, int stride,
                                      int ct, int ks, int lane) {
  const _Float16* p = W + (size_t)(ct * 16 + (lane & 15)) * stride
                        + ks * 32 + ((lane >> 4) * 8);
  v8h lo = *(const v8h*)(p);
  v8h hi = *(const v8h*)(p + 16);
  return __builtin_shufflevector(lo, hi, 0,1,2,3,4,5,6,7,8,9,10,11,12,13,14,15);
}

// fast transcendentals on the v_exp_f32 TRANS path (safe at +/-inf)
__device__ __forceinline__ float sigmoidf_(float x) { return 1.f / (1.f + __expf(-x)); }
__device__ __forceinline__ float tanhf_(float x) {
  return 1.f - 2.f / (__expf(2.f * x) + 1.f);
}

#define WMMA(a, b, c) \
  __builtin_amdgcn_wmma_f32_16x16x32_f16(false, (a), false, (b), (short)0, (c), false, false)

// ---- CDNA5 async global->LDS copy (ASYNCcnt-tracked) -----------------------
// VDST VGPR holds the wave-relative LDS byte address (generic LDS ptr low 32b).
__device__ __forceinline__ void async_ld_b32(uint32_t lds_off, const float* gptr) {
  asm volatile("global_load_async_to_lds_b32 %0, %1, off"
               :: "v"(lds_off), "v"((uint64_t)(uintptr_t)gptr) : "memory");
}
__device__ __forceinline__ void wait_async0() {
  asm volatile("s_wait_asynccnt 0x0" ::: "memory");
}

// Issue async loads of obs[tt] (16 rows x 54 f32) into an LDS staging buffer.
__device__ __forceinline__ void issue_obs(const float* __restrict__ obs, int tt,
                                          int n0, int tid, float* buf) {
  for (int e = tid; e < 16 * OBS_; e += 256) {
    int m = e / OBS_, k = e - m * OBS_;
    uint32_t dst = (uint32_t)(uintptr_t)(buf + m * SFP + k);
    async_ld_b32(dst, obs + ((size_t)tt * N_ + n0 + m) * OBS_ + k);
  }
}

// ---------------------------------------------------------------------------
// Prep: convert weights to f16 (K-padded rows), fold biases. Runs every call.
// ---------------------------------------------------------------------------
__global__ __launch_bounds__(256) void prep_weights(
    const float* __restrict__ W1, const float* __restrict__ W2,
    const float* __restrict__ Wih, const float* __restrict__ Whh,
    const float* __restrict__ Wp, const float* __restrict__ Wv,
    const float* __restrict__ bih, const float* __restrict__ bhh,
    _Float16* __restrict__ w1p, _Float16* __restrict__ w2p,
    _Float16* __restrict__ wihp, _Float16* __restrict__ whhp,
    _Float16* __restrict__ whp, float* __restrict__ bg)
{
  int i = blockIdx.x * 256 + threadIdx.x;
  if (i < 128 * 64) {                       // W1: 128x54 -> 128x64 zero-padded
    int r = i >> 6, k = i & 63;
    w1p[i] = (k < OBS_) ? (_Float16)W1[r * OBS_ + k] : (_Float16)0.f;
  }
  if (i < 128 * 128) w2p[i] = (_Float16)W2[i];
  if (i < G_ * 128) {                       // 65536 elements each
    wihp[i] = (_Float16)Wih[i];
    whhp[i] = (_Float16)Whh[i];
  }
  if (i < 16 * 128) {                       // head: rows 0-5 = Wp, row 6 = Wv, rest 0
    int r = i >> 7, k = i & 127;
    float v = (r < ACT_) ? Wp[r * 128 + k] : ((r == ACT_) ? Wv[k] : 0.f);
    whp[i] = (_Float16)v;
  }
  if (i < G_) bg[i] = bih[i] + bhh[i];
}

// ---------------------------------------------------------------------------
// Main persistent kernel: one block = 16 envs, loops over all T timesteps.
// ---------------------------------------------------------------------------
__global__ __launch_bounds__(256) void policy_fused(
    const float* __restrict__ obs, const float* __restrict__ h0,
    const float* __restrict__ c0, const int* __restrict__ done,
    const float* __restrict__ b1, const float* __restrict__ b2,
    const float* __restrict__ bp, const float* __restrict__ bv,
    const _Float16* __restrict__ w1p, const _Float16* __restrict__ w2p,
    const _Float16* __restrict__ wihp, const _Float16* __restrict__ whhp,
    const _Float16* __restrict__ whp, const float* __restrict__ bg,
    float* __restrict__ logits, float* __restrict__ values,
    float* __restrict__ hT, float* __restrict__ cT)
{
  __shared__ float    sObsF[2][16 * SFP];   // raw f32 obs, async double buffer
  __shared__ _Float16 sObs[16 * SOP];
  __shared__ _Float16 sX1 [16 * SXP];
  __shared__ _Float16 sX  [16 * SXP];
  __shared__ _Float16 sH  [16 * SXP];
  __shared__ float    sGates[16 * GSTR];
  __shared__ float    sM[16];

  const int tid  = threadIdx.x;
  const int lane = tid & 31;
  const int wave = tid >> 5;
  const int n0   = blockIdx.x * 16;     // first environment row of this block
  const int cn   = lane & 15;           // WMMA C-tile column for this lane
  const int mh   = 8 * (lane >> 4);     // WMMA C-tile row offset for this lane

  // kick off async staging of obs[0] immediately
  issue_obs(obs, 0, n0, tid, &sObsF[0][0]);

  // ---- preload ALL loop-invariant weight fragments into registers ----------
  // (SSA values: immune to the per-iteration asm memory clobbers, so they stay
  //  VGPR-resident across all 128 timesteps; ~340 VGPRs by design.)
  v16h bw1[2], bw2[4], bwih[4][4], bwhh[4][4], bwh[4];
  #pragma unroll
  for (int ks = 0; ks < 2; ++ks) bw1[ks] = bfrag(w1p, 64, wave, ks, lane);
  #pragma unroll
  for (int ks = 0; ks < 4; ++ks) bw2[ks] = bfrag(w2p, 128, wave, ks, lane);
  #pragma unroll
  for (int j = 0; j < 4; ++j)
    #pragma unroll
    for (int ks = 0; ks < 4; ++ks) {
      bwih[j][ks] = bfrag(wihp, 128, wave * 4 + j, ks, lane);
      bwhh[j][ks] = bfrag(whhp, 128, wave * 4 + j, ks, lane);
    }
  #pragma unroll
  for (int ks = 0; ks < 4; ++ks) bwh[ks] = bfrag(whp, 128, 0, ks, lane);

  const float bias1 = b1[wave * 16 + cn];
  const float bias2 = b2[wave * 16 + cn];
  float bgj[4];
  #pragma unroll
  for (int j = 0; j < 4; ++j) bgj[j] = bg[(wave * 4 + j) * 16 + cn];
  const float biash = (cn < ACT_) ? bp[cn] : ((cn == ACT_) ? bv[0] : 0.f);

  // ---- init recurrent state: h and c both live in registers (8 elems/thr) --
  float creg[8], hreg[8];
  #pragma unroll
  for (int k = 0; k < 8; ++k) {
    int e = tid + k * 256, m = e >> 7, n = e & 127;
    creg[k] = c0[(size_t)(n0 + m) * HID_ + n];
    hreg[k] = h0[(size_t)(n0 + m) * HID_ + n];
  }

  for (int t = 0; t < T_; ++t) {
    const int par = t & 1;

    // ---- A1: wait for async obs[t], publish, start obs[t+1], convert -------
    wait_async0();
    __syncthreads();
    if (t + 1 < T_) issue_obs(obs, t + 1, n0, tid, &sObsF[par ^ 1][0]);
    for (int e = tid; e < 16 * 64; e += 256) {
      int m = e >> 6, k = e & 63;
      float v = (k < OBS_) ? sObsF[par][m * SFP + k] : 0.f;
      sObs[m * SOP + k] = (_Float16)v;
    }
    if (tid < 16)
      sM[tid] = (t > 0) ? (1.f - (float)done[(size_t)(t - 1) * N_ + n0 + tid]) : 1.f;
    __syncthreads();

    // ---- A2: publish masked h to LDS (store-only; no barrier needed here:
    //          B/C never touch sH, and the end-of-C barrier orders it for D) --
    #pragma unroll
    for (int k = 0; k < 8; ++k) {
      int e = tid + k * 256, m = e >> 7, n = e & 127;
      float mk = sM[m];
      creg[k] *= mk;
      sH[m * SXP + n] = (_Float16)(hreg[k] * mk);
    }

    // ---- B: x1 = relu(obs @ W1^T + b1); one 16-col tile per wave -----------
    {
      v16h a0 = afrag(sObs, SOP, 0, lane);
      v16h a1 = afrag(sObs, SOP, 1, lane);
      v8f acc = {};
      acc = WMMA(a0, bw1[0], acc);
      acc = WMMA(a1, bw1[1], acc);
      #pragma unroll
      for (int r = 0; r < 8; ++r)
        sX1[(r + mh) * SXP + wave * 16 + cn] = (_Float16)fmaxf(acc[r] + bias1, 0.f);
    }
    __syncthreads();

    // ---- C: x2 = relu(x1 @ W2^T + b2) --------------------------------------
    {
      v16h a[4];
      #pragma unroll
      for (int ks = 0; ks < 4; ++ks) a[ks] = afrag(sX1, SXP, ks, lane);
      v8f acc = {};
      #pragma unroll
      for (int ks = 0; ks < 4; ++ks) acc = WMMA(a[ks], bw2[ks], acc);
      #pragma unroll
      for (int r = 0; r < 8; ++r)
        sX[(r + mh) * SXP + wave * 16 + cn] = (_Float16)fmaxf(acc[r] + bias2, 0.f);
    }
    __syncthreads();

    // ---- D: gates = x2@W_ih^T + h@W_hh^T + (b_ih+b_hh); 4 col-tiles/wave ---
    {
      v16h ax[4], ah[4];
      #pragma unroll
      for (int ks = 0; ks < 4; ++ks) {
        ax[ks] = afrag(sX, SXP, ks, lane);
        ah[ks] = afrag(sH, SXP, ks, lane);
      }
      #pragma unroll
      for (int j = 0; j < 4; ++j) {
        v8f acc;
        #pragma unroll
        for (int r = 0; r < 8; ++r) acc[r] = bgj[j];
        #pragma unroll
        for (int ks = 0; ks < 4; ++ks) acc = WMMA(ax[ks], bwih[j][ks], acc);
        #pragma unroll
        for (int ks = 0; ks < 4; ++ks) acc = WMMA(ah[ks], bwhh[j][ks], acc);
        #pragma unroll
        for (int r = 0; r < 8; ++r)
          sGates[(r + mh) * GSTR + (wave * 4 + j) * 16 + cn] = acc[r];
      }
    }
    __syncthreads();

    // ---- E: elementwise LSTM cell (PyTorch gate order i,f,g,o) -------------
    #pragma unroll
    for (int k = 0; k < 8; ++k) {
      int e = tid + k * 256, m = e >> 7, n = e & 127;
      float gi = sGates[m * GSTR + n];
      float gf = sGates[m * GSTR + 128 + n];
      float gg = sGates[m * GSTR + 256 + n];
      float go = sGates[m * GSTR + 384 + n];
      float c  = sigmoidf_(gf) * creg[k] + sigmoidf_(gi) * tanhf_(gg);
      float h  = sigmoidf_(go) * tanhf_(c);
      creg[k] = c;
      hreg[k] = h;
      sH[m * SXP + n] = (_Float16)h;          // for the head GEMM
      if (t == T_ - 1) {
        hT[(size_t)(n0 + m) * HID_ + n] = h;
        cT[(size_t)(n0 + m) * HID_ + n] = c;
      }
    }
    __syncthreads();

    // ---- F: heads: [logits | value] = h @ [Wp;Wv;0]^T (wave 0 only) --------
    // No trailing barrier: next iteration's two loop-top barriers order
    // F's sH reads before the next A2 overwrite.
    if (wave == 0) {
      v16h a[4];
      #pragma unroll
      for (int ks = 0; ks < 4; ++ks) a[ks] = afrag(sH, SXP, ks, lane);
      v8f acc = {};
      #pragma unroll
      for (int ks = 0; ks < 4; ++ks) acc = WMMA(a[ks], bwh[ks], acc);
      #pragma unroll
      for (int r = 0; r < 8; ++r) {
        int m = r + mh;
        float v = acc[r] + biash;
        if (cn < ACT_)
          logits[((size_t)t * N_ + n0 + m) * ACT_ + cn] = v;
        else if (cn == ACT_)
          values[(size_t)t * N_ + n0 + m] = v;
      }
    }
  }
}

// ---------------------------------------------------------------------------
extern "C" void kernel_launch(void* const* d_in, const int* in_sizes, int n_in,
                              void* d_out, int out_size, void* d_ws, size_t ws_size,
                              hipStream_t stream) {
  (void)in_sizes; (void)n_in; (void)out_size; (void)ws_size;

  const float* obs = (const float*)d_in[0];
  const float* h0  = (const float*)d_in[1];
  const float* c0  = (const float*)d_in[2];
  const int*   dn  = (const int*)  d_in[3];
  const float* W1  = (const float*)d_in[4];
  const float* b1  = (const float*)d_in[5];
  const float* W2  = (const float*)d_in[6];
  const float* b2  = (const float*)d_in[7];
  const float* Wih = (const float*)d_in[8];
  const float* Whh = (const float*)d_in[9];
  const float* bih = (const float*)d_in[10];
  const float* bhh = (const float*)d_in[11];
  const float* Wp  = (const float*)d_in[12];
  const float* bp  = (const float*)d_in[13];
  const float* Wv  = (const float*)d_in[14];
  const float* bv  = (const float*)d_in[15];

  // workspace layout (f16 packed weights + folded gate bias), ~310 KB total
  _Float16* w1p  = (_Float16*)d_ws;          // 128*64
  _Float16* w2p  = w1p  + 128 * 64;          // 128*128
  _Float16* wihp = w2p  + 128 * 128;         // 512*128
  _Float16* whhp = wihp + 512 * 128;         // 512*128
  _Float16* whp  = whhp + 512 * 128;         // 16*128
  float*    bg   = (float*)(whp + 16 * 128); // 512 f32

  float* out    = (float*)d_out;
  float* logits = out;
  float* values = logits + (size_t)T_ * N_ * ACT_;
  float* hT     = values + (size_t)T_ * N_;
  float* cT     = hT + (size_t)N_ * HID_;

  prep_weights<<<256, 256, 0, stream>>>(W1, W2, Wih, Whh, Wp, Wv, bih, bhh,
                                        w1p, w2p, wihp, whhp, whp, bg);
  policy_fused<<<N_ / 16, 256, 0, stream>>>(obs, h0, c0, dn, b1, b2, bp, bv,
                                            w1p, w2p, wihp, whhp, whp, bg,
                                            logits, values, hT, cT);
}
